// LSTM_21878563406096
// MI455X (gfx1250) — matrix operand order repaired and checked by reading, not compiled, not run on hardware
//
#include <hip/hip_runtime.h>

typedef __attribute__((ext_vector_type(16))) __bf16 v16bf;
typedef __attribute__((ext_vector_type(8)))  __bf16 v8bf;
typedef __attribute__((ext_vector_type(8)))  float  v8f;

#define BATCH  128
#define TLEN   256
#define HID    128
#define WIN    16
#define NSTEPS 3840   // 240 windows * 16

// ---- fast activations: hardware v_tanh_f32 on gfx1250 (single TRANS op) ----
__device__ __forceinline__ float fast_tanh(float z) {
#if __has_builtin(__builtin_amdgcn_tanhf)
    return __builtin_amdgcn_tanhf(z);
#else
    // fallback: exp + v_rcp_f32 (no IEEE-div fixup sequence)
    return 1.0f - 2.0f * __builtin_amdgcn_rcpf(__expf(2.0f * z) + 1.0f);
#endif
}
__device__ __forceinline__ float fast_sigmoid(float z) {
    return __builtin_fmaf(0.5f, fast_tanh(0.5f * z), 0.5f);
}

// 8 workgroups x 256 threads (8 waves). WG g owns batch rows 16g..16g+15.
// Wave w owns hidden units 16w..16w+15 (its 4 gate tiles i/f/g/o).
// Lane l: n = l&15 (gate column / hidden unit), hi = l>>4 selects M/K halves.
__launch_bounds__(256)
__global__ void lstm_wmma_kernel(const float* __restrict__ x,
                                 const float* __restrict__ W_ih,
                                 const float* __restrict__ W_hh,
                                 const float* __restrict__ b_ih,
                                 const float* __restrict__ b_hh,
                                 const float* __restrict__ fc_W,
                                 const float* __restrict__ fc_b,
                                 float* __restrict__ out)
{
    __shared__ float x_tile[16 * TLEN];                      // 16 KB: this WG's x rows
    __shared__ __align__(16) __bf16 hbuf[2][16 * HID];       // 2 x 4 KB: ping-pong h tiles
    __shared__ float fcacc[16];                              // FC dot-product accumulators

    const int tid  = threadIdx.x;
    const int wave = tid >> 5;        // 0..7
    const int lane = tid & 31;
    const int hi   = lane >> 4;       // 0/1: which half of lanes
    const int ln   = lane & 15;
    const int m0   = blockIdx.x * 16; // first batch row of this WG

    // ---- init LDS ----
    for (int i = tid; i < 16 * TLEN; i += 256)
        x_tile[i] = x[m0 * TLEN + i];
    for (int i = tid; i < 16 * HID; i += 256)
        hbuf[0][i] = (__bf16)0.0f;
    if (tid < 16) fcacc[tid] = 0.0f;

    // out[:, 0:16] = x[:, 0:16]
    {
        const int r = tid >> 4, col = tid & 15;
        out[(m0 + r) * TLEN + col] = x[(m0 + r) * TLEN + col];
    }

    // ---- per-lane constants (this lane's gate column = hidden unit `unit`) ----
    const int unit = wave * 16 + ln;              // 0..127
    float wih[4], bias[4];
    #pragma unroll
    for (int q = 0; q < 4; ++q) {
        const int j = q * HID + unit;
        wih[q]  = W_ih[j];
        bias[q] = b_ih[j] + b_hh[j];
    }
    const float fcw = fc_W[unit];
    const float fcb = fc_b[0];

    // ---- B fragments: W_hh tiles in bf16, resident in registers ----
    // B is K x N (32 x 16) per (gate q, k-chunk kk): B[k][n] = W_hh[q*H + 16w + n][k]
    // lane layout: n = ln; element e holds k = kk*32 + hi*16 + e  (contiguous row read)
    v16bf bfrag[4][4];
    #pragma unroll
    for (int q = 0; q < 4; ++q) {
        const float* wrow = W_hh + (q * HID + unit) * HID;
        #pragma unroll
        for (int kk = 0; kk < 4; ++kk) {
            const int kb = kk * 32 + hi * 16;
            v16bf f;
            #pragma unroll
            for (int e = 0; e < 16; ++e)
                f[e] = (__bf16)wrow[kb + e];
            bfrag[q][kk] = f;
        }
    }

    // ---- persistent cell state: c for (m = r + 8*hi, n = unit) ----
    float cst[8];
    #pragma unroll
    for (int r = 0; r < 8; ++r) cst[r] = 0.0f;

    __syncthreads();

    for (int step = 0; step < NSTEPS; ++step) {
        const int s    = step >> 4;
        const int t    = step & 15;
        const int time = s + t;                 // sliding-window time index
        const __bf16* hRd = hbuf[step & 1];
        __bf16*       hWr = hbuf[(step & 1) ^ 1];

        // ---- A fragments from LDS h tile (16x32 bf16 per kk) ----
        // lane: m = ln; elem e -> k = kk*32 + 16*hi + e (contiguous)
        v16bf afrag[4];
        #pragma unroll
        for (int kk = 0; kk < 4; ++kk) {
            const __bf16* hrow = hRd + ln * HID + kk * 32 + hi * 16;
            const v8bf a0 = *(const v8bf*)(hrow);
            const v8bf a1 = *(const v8bf*)(hrow + 8);
            v16bf f;
            #pragma unroll
            for (int e = 0; e < 8; ++e) { f[e] = a0[e]; f[e + 8] = a1[e]; }
            afrag[kk] = f;
        }

        // ---- x contribution folded into WMMA C operand ----
        float xv[8];
        #pragma unroll
        for (int r = 0; r < 8; ++r)
            xv[r] = x_tile[(r + 8 * hi) * TLEN + time];

        // ---- gates = x*W_ih + b + h @ W_hh^T : 4 chained WMMAs per gate tile ----
        v8f acc[4];
        #pragma unroll
        for (int q = 0; q < 4; ++q) {
            v8f a;
            #pragma unroll
            for (int r = 0; r < 8; ++r)
                a[r] = __builtin_fmaf(xv[r], wih[q], bias[q]);
            #pragma unroll
            for (int kk = 0; kk < 4; ++kk)
                a = __builtin_amdgcn_wmma_f32_16x16x32_bf16(
                        false, afrag[kk], false, bfrag[q][kk],
                        (short)0, a, false, false);
            acc[q] = a;
        }

        // ---- LSTM cell update (lane owns (8 batch rows) x (1 hidden unit)) ----
        #pragma unroll
        for (int r = 0; r < 8; ++r) {
            const int   m  = r + 8 * hi;
            const float si = fast_sigmoid(acc[0][r]);
            const float sf = fast_sigmoid(acc[1][r]);
            const float tg = fast_tanh(acc[2][r]);
            const float so = fast_sigmoid(acc[3][r]);
            const float cn = sf * cst[r] + si * tg;
            cst[r] = cn;
            const float hv = so * fast_tanh(cn);
            hWr[m * HID + unit] = (__bf16)hv;           // h for next step (other buffer)
            if (t == WIN - 1)
                atomicAdd(&fcacc[m], hv * fcw);         // ds_add_f32 LDS atomic
        }
        __syncthreads();   // new h visible; FC atomics complete

        if (t == WIN - 1 && tid < 16) {
            const float v = fcacc[tid] + fcb;
            out[(m0 + tid) * TLEN + WIN + s] = (v > 0.0f) ? v : 0.3f * v;
            fcacc[tid] = 0.0f;   // next window's adds are >=15 barriers away
        }
    }
}

extern "C" void kernel_launch(void* const* d_in, const int* in_sizes, int n_in,
                              void* d_out, int out_size, void* d_ws, size_t ws_size,
                              hipStream_t stream)
{
    const float* x    = (const float*)d_in[0];
    const float* W_ih = (const float*)d_in[1];
    const float* W_hh = (const float*)d_in[2];
    const float* b_ih = (const float*)d_in[3];
    const float* b_hh = (const float*)d_in[4];
    const float* fc_W = (const float*)d_in[5];
    const float* fc_b = (const float*)d_in[6];
    float* out = (float*)d_out;

    lstm_wmma_kernel<<<dim3(BATCH / 16), dim3(256), 0, stream>>>(
        x, W_ih, W_hh, b_ih, b_hh, fc_W, fc_b, out);
}
